// QuantLayerMoDiff_75943611728160
// MI455X (gfx1250) — 8-bit matrix operand order repaired and checked by reading, compiled, not run on hardware
//
#include <hip/hip_runtime.h>


#define NB     16
#define CIN    320
#define COUT   320
#define HH     64
#define WW     64
#define PH     66
#define PW     66
#define NPIX   (NB * HH * WW)
#define NELEM  (NB * CIN * HH * WW)
#define NPAD   (NB * PH * PW * CIN)
#define KCH    5
#define NT     20
#define NSTEP  (9 * KCH)
#define QMAXF  127.0f

typedef __attribute__((ext_vector_type(8))) int v8i;

__global__ void k_init(unsigned* g) { *g = 0u; }

__global__ __launch_bounds__(256) void k_zero(int4* __restrict__ p, int n16) {
    const int i = blockIdx.x * 256 + threadIdx.x;
    if (i < n16) p[i] = make_int4(0, 0, 0, 0);
}

__global__ __launch_bounds__(256) void k_wscale(const float* __restrict__ wgt,
                                                float* __restrict__ wscale) {
    const int co  = blockIdx.x;
    const int tid = threadIdx.x;
    __shared__ float red[256];
    float m = 0.0f;
    const float* wrow = wgt + (size_t)co * (CIN * 9);
    for (int j = tid; j < CIN * 9; j += 256) m = fmaxf(m, fabsf(wrow[j]));
    red[tid] = m;
    __syncthreads();
    for (int s = 128; s > 0; s >>= 1) {
        if (tid < s) red[tid] = fmaxf(red[tid], red[tid + s]);
        __syncthreads();
    }
    if (tid == 0) wscale[co] = fmaxf(red[0] * (1.0f / QMAXF), 1e-8f);
}

__global__ __launch_bounds__(256) void k_absmax(const float* __restrict__ x,
                                                const float* __restrict__ ah,
                                                unsigned* __restrict__ g, int n) {
    const int tid = threadIdx.x;
    float m = 0.0f;
    for (int i = blockIdx.x * 256 + tid; i < n; i += gridDim.x * 256)
        m = fmaxf(m, fabsf(x[i] - ah[i]));
    __shared__ float red[256];
    red[tid] = m;
    __syncthreads();
    for (int s = 128; s > 0; s >>= 1) {
        if (tid < s) red[tid] = fmaxf(red[tid], red[tid + s]);
        __syncthreads();
    }
    if (tid == 0) atomicMax(g, __float_as_uint(red[0]));
}

__global__ __launch_bounds__(256) void k_quant(const float* __restrict__ x,
                                               const float* __restrict__ ah,
                                               const float* __restrict__ absmax,
                                               signed char* __restrict__ qres,
                                               float* __restrict__ ahat_new) {
    const int i = blockIdx.x * 256 + threadIdx.x;
    if (i >= NELEM) return;
    const float a_scale = fmaxf(absmax[0] * (1.0f / QMAXF), 1e-8f);
    const float av = ah[i];
    const float r  = x[i] - av;
    const float q  = fminf(fmaxf(rintf(r / a_scale), -QMAXF), QMAXF);
    const int w = i & 63;
    const int h = (i >> 6) & 63;
    const int t = i >> 12;
    const int c = t % CIN;
    const int b = t / CIN;
    qres[((size_t)((b * PH + h + 1) * PW + w + 1)) * CIN + c] = (signed char)q;
    ahat_new[i] = av + q * a_scale;
}

__global__ __launch_bounds__(256) void k_pack(const float* __restrict__ wgt,
                                              const float* __restrict__ wscale,
                                              signed char* __restrict__ wpack) {
    const int idx = blockIdx.x * 256 + threadIdx.x;
    const int bt   = idx & 3;
    const int v    = (idx >> 2) & 7;
    const int lane = (idx >> 5) & 31;
    const int tile = idx >> 10;
    const int nt   = tile % NT;
    const int kc   = (tile / NT) % KCH;
    const int khw  = tile / (NT * KCH);
    const int kh = khw / 3, kw = khw % 3;
    const int n = lane & 15, half = lane >> 4;
    const int K = ((v >> 1) << 4) + (half << 3) + ((v & 1) << 2) + bt;
    const int cout = nt * 16 + n;
    const int ci   = kc * 64 + K;
    const float s = wscale[cout];
    const float w = wgt[((size_t)(cout * CIN + ci) * 3 + kh) * 3 + kw];
    const float q = fminf(fmaxf(rintf(w / s), -QMAXF), QMAXF);
    wpack[idx] = (signed char)q;
}

__global__ __launch_bounds__(256, 2) void k_conv(
        const signed char* __restrict__ qres,
        const signed char* __restrict__ wpack,
        const float* __restrict__ wscale,
        const float* __restrict__ absmax,
        const float* __restrict__ ohat,
        float* __restrict__ out) {
    const int lane = threadIdx.x & 31;
    const int wid  = blockIdx.x * 8 + (threadIdx.x >> 5);
    const int ng   = wid % 5;
    const int pt   = wid / 5;
    const int b    = pt >> 8;
    const int rem  = pt & 255;
    const int h    = rem >> 2;
    const int w0   = (rem & 3) << 4;
    const int M    = lane & 15;
    const int half = lane >> 4;

    const float a_scale = fmaxf(absmax[0] * (1.0f / QMAXF), 1e-8f);

    const signed char* apix =
        qres + ((ptrdiff_t)(b * PH + h) * PW + (w0 + M)) * CIN + half * 8;
    const signed char* bpack0 =
        wpack + (ptrdiff_t)(ng * 4) * 1024 + lane * 32;

    auto loadA = [&](int s) -> v8i {
        const int khw = s / KCH, kc = s % KCH;
        const int kh = khw / 3, kw = khw % 3;
        const signed char* p = apix + (ptrdiff_t)(kh * PW + kw) * CIN + kc * 64;
        const int2 c0 = *(const int2*)(p);
        const int2 c1 = *(const int2*)(p + 16);
        const int2 c2 = *(const int2*)(p + 32);
        const int2 c3 = *(const int2*)(p + 48);
        v8i a;
        a[0] = c0.x; a[1] = c0.y;
        a[2] = c1.x; a[3] = c1.y;
        a[4] = c2.x; a[5] = c2.y;
        a[6] = c3.x; a[7] = c3.y;
        return a;
    };
    auto loadB = [&](int s, int j) -> v8i {
        const signed char* bt = bpack0 + (ptrdiff_t)(s * NT + j) * 1024;
        return *(const v8i*)(bt);
    };

    const v8i vzero = {0, 0, 0, 0, 0, 0, 0, 0};
    v8i acc0 = vzero, acc1 = vzero, acc2 = vzero, acc3 = vzero;

    v8i aC  = loadA(0);
    v8i b0C = loadB(0, 0), b1C = loadB(0, 1), b2C = loadB(0, 2), b3C = loadB(0, 3);

#pragma unroll
    for (int s = 0; s < NSTEP - 1; ++s) {
        if ((s % KCH) == 0)
            __builtin_prefetch(bpack0 + (ptrdiff_t)((s + KCH) * NT) * 1024, 0, 1);

        const v8i aN  = loadA(s + 1);
        const v8i b0N = loadB(s + 1, 0);
        const v8i b1N = loadB(s + 1, 1);
        const v8i b2N = loadB(s + 1, 2);
        const v8i b3N = loadB(s + 1, 3);

        acc0 = __builtin_amdgcn_wmma_i32_16x16x64_iu8(true, aC, true, b0C,
                                                      acc0, false, false);
        acc1 = __builtin_amdgcn_wmma_i32_16x16x64_iu8(true, aC, true, b1C,
                                                      acc1, false, false);
        acc2 = __builtin_amdgcn_wmma_i32_16x16x64_iu8(true, aC, true, b2C,
                                                      acc2, false, false);
        acc3 = __builtin_amdgcn_wmma_i32_16x16x64_iu8(true, aC, true, b3C,
                                                      acc3, false, false);

        aC = aN; b0C = b0N; b1C = b1N; b2C = b2N; b3C = b3N;
    }
    acc0 = __builtin_amdgcn_wmma_i32_16x16x64_iu8(true, aC, true, b0C,
                                                  acc0, false, false);
    acc1 = __builtin_amdgcn_wmma_i32_16x16x64_iu8(true, aC, true, b1C,
                                                  acc1, false, false);
    acc2 = __builtin_amdgcn_wmma_i32_16x16x64_iu8(true, aC, true, b2C,
                                                  acc2, false, false);
    acc3 = __builtin_amdgcn_wmma_i32_16x16x64_iu8(true, aC, true, b3C,
                                                  acc3, false, false);

    auto epi = [&](int j, const v8i& av) {
        const int cout = ng * 64 + j * 16 + (lane & 15);
        const float sc = a_scale * wscale[cout];
        const size_t base =
            ((size_t)(b * COUT + cout) * HH + h) * WW + w0 + 8 * half;
        const float4 o0 = *(const float4*)(ohat + base);
        const float4 o1 = *(const float4*)(ohat + base + 4);
        float4 r0, r1;
        r0.x = o0.x + sc * (float)av[0];
        r0.y = o0.y + sc * (float)av[1];
        r0.z = o0.z + sc * (float)av[2];
        r0.w = o0.w + sc * (float)av[3];
        r1.x = o1.x + sc * (float)av[4];
        r1.y = o1.y + sc * (float)av[5];
        r1.z = o1.z + sc * (float)av[6];
        r1.w = o1.w + sc * (float)av[7];
        *(float4*)(out + base)     = r0;
        *(float4*)(out + base + 4) = r1;
    };
    epi(0, acc0);
    epi(1, acc1);
    epi(2, acc2);
    epi(3, acc3);
}

extern "C" void kernel_launch(void* const* d_in, const int* in_sizes, int n_in,
                              void* d_out, int out_size, void* d_ws, size_t ws_size,
                              hipStream_t stream) {
    const float* x   = (const float*)d_in[0];
    const float* wgt = (const float*)d_in[1];
    const float* ah  = (const float*)d_in[2];
    const float* oh  = (const float*)d_in[3];

    float* out      = (float*)d_out;
    float* ahat_new = out + (size_t)NELEM;

    char* ws = (char*)d_ws;
    unsigned*    absmax = (unsigned*)ws;
    float*       wscale = (float*)(ws + 128);
    signed char* wpack  = (signed char*)(ws + 2048);
    signed char* qres   = (signed char*)(ws + (1 << 20));

    k_init<<<1, 1, 0, stream>>>(absmax);
    k_zero<<<NPAD / 16 / 256, 256, 0, stream>>>((int4*)qres, NPAD / 16);
    k_wscale<<<COUT, 256, 0, stream>>>(wgt, wscale);
    k_absmax<<<4096, 256, 0, stream>>>(x, ah, absmax, NELEM);
    k_quant<<<NELEM / 256, 256, 0, stream>>>(x, ah, (const float*)absmax,
                                             qres, ahat_new);
    k_pack<<<(9 * KCH * NT * 1024) / 256, 256, 0, stream>>>(wgt, wscale, wpack);
    k_conv<<<(NPIX / 16) * KCH / 8, 256, 0, stream>>>(
        qres, wpack, wscale, (const float*)absmax, oh, out);
}
